// Attention_79104707657812
// MI455X (gfx1250) — matrix operand order repaired and checked by reading, not compiled, not run on hardware
//
#include <hip/hip_runtime.h>

typedef __attribute__((ext_vector_type(16))) _Float16 v16h;
typedef __attribute__((ext_vector_type(8)))  _Float16 v8h;
typedef __attribute__((ext_vector_type(8)))  float    v8f;

#define WMMA_F32_F16(a, b, c) \
  __builtin_amdgcn_wmma_f32_16x16x32_f16(false, (a), false, (b), (short)0, (c), false, false)

static constexpr size_t OUT_K_OFF = 4ull * 1024 * 1024;                 // present[0]=k offset (floats)
static constexpr size_t OUT_V_OFF = OUT_K_OFF + 4ull * 16 * 1024 * 64;  // present[1]=v offset

// ---------- workspace layout (bytes, all >=128B aligned) ----------
static constexpr size_t WS_WAT = 0;                                   // 3072x1024 f16 (w_attn^T)
static constexpr size_t WS_WPT = WS_WAT + 3072ull * 1024 * 2;         // 1024x1024 f16 (w_proj^T)
static constexpr size_t WS_QH  = WS_WPT + 1024ull * 1024 * 2;         // (B,NH,S,HD) f16
static constexpr size_t WS_KH  = WS_QH + 4ull * 16 * 1024 * 64 * 2;   // (B,NH,S,HD) f16
static constexpr size_t WS_VT  = WS_KH + 4ull * 16 * 1024 * 64 * 2;   // (B,NH,HD,S) f16
static constexpr size_t WS_AH  = WS_VT + 4ull * 16 * 1024 * 64 * 2;   // (B*S, NX) f16
static constexpr size_t WS_XH  = WS_AH + 4096ull * 1024 * 2;          // (B*S, NX) f16 copy of x

// ---------- fragment load helpers (M/N lanes per CDNA5 ISA 7.12.2; contiguous K order in both A and B) ----------
// A fragment (16x32 f16): M = lane&15, kb = 8*(lane>>4); elem e -> K=k0+2*kb+e
// (16 contiguous K per lane half, matching load_b16).
__device__ __forceinline__ v16h load_a_f16(const _Float16* rowp, int kb, int k0) {
  const v8h lo = *reinterpret_cast<const v8h*>(rowp + k0 + 2 * kb);
  const v8h hi = *reinterpret_cast<const v8h*>(rowp + k0 + 2 * kb + 8);
  v16h r;
#pragma unroll
  for (int i = 0; i < 8; ++i) { r[i] = lo[i]; r[i + 8] = hi[i]; }
  return r;
}

// B fragment (32x16 f16): N = lane&15, K = 16*(lane>>4) + e -> 16 contiguous f16.
__device__ __forceinline__ v16h load_b16(const _Float16* p) {
  return *reinterpret_cast<const v16h*>(p);
}

__device__ __forceinline__ void zero8(v8f& v) {
#pragma unroll
  for (int i = 0; i < 8; ++i) v[i] = 0.0f;
}

// Async copy of one 32-row x 64-col f16 K-tile (4KB) into LDS.
// Lane L copies row kb0+L (128B) as 8 x b128 chunks. Tracked by ASYNCcnt.
__device__ __forceinline__ void async_ktile(const _Float16* kbh, int kb0, int lane,
                                            unsigned lds_base) {
  const _Float16* g = kbh + (size_t)(kb0 + lane) * 64;
  const unsigned l = lds_base + (unsigned)lane * 128u;
#pragma unroll
  for (int i = 0; i < 8; ++i) {
    asm volatile("global_load_async_to_lds_b128 %0, %1, off"
                 :: "v"(l + (unsigned)(i * 16)), "v"(g + i * 8) : "memory");
  }
}

// ---------- kernel 0: convert weights (transposed) and x to f16 ----------
__global__ __launch_bounds__(256) void cvt_kernel(const float* __restrict__ x,
                                                  const float* __restrict__ w_attn,
                                                  const float* __restrict__ w_proj,
                                                  _Float16* __restrict__ xh,
                                                  _Float16* __restrict__ waT,
                                                  _Float16* __restrict__ wpT) {
  const size_t i = (size_t)blockIdx.x * 256 + threadIdx.x;
  const size_t NA = 3072ull * 1024;            // waT
  const size_t NB = NA + 1024ull * 1024;       // wpT
  if (i < NA) {
    int n = (int)(i >> 10), k = (int)(i & 1023);
    waT[i] = (_Float16)w_attn[(size_t)k * 3072 + n];
  } else if (i < NB) {
    size_t j = i - NA;
    int n = (int)(j >> 10), k = (int)(j & 1023);
    wpT[j] = (_Float16)w_proj[(size_t)k * 1024 + n];
  } else {
    size_t j = i - NB;                         // < 4096*1024
    xh[j] = (_Float16)x[j];
  }
}

// ---------- kernel 1: QKV GEMM, 16x64 per wave, routed epilogue ----------
__global__ __launch_bounds__(256) void qkv_kernel(const _Float16* __restrict__ xh,
                                                  const _Float16* __restrict__ waT,
                                                  const float* __restrict__ b_attn,
                                                  _Float16* __restrict__ qh,
                                                  _Float16* __restrict__ kh,
                                                  _Float16* __restrict__ vT,
                                                  float* __restrict__ outp) {
  const int wave = threadIdx.x >> 5;
  const int lane = threadIdx.x & 31;
  const int gw = blockIdx.x * 8 + wave;   // over 256 m-tiles x 48 n-groups
  const int NG = 3072 / 64;
  const int tm = gw / NG, tg = gw % NG;
  const int m0 = tm * 16, n0 = tg * 64;
  const int row = lane & 15;
  const int h16 = lane >> 4;
  const int kb = 8 * h16;

  const _Float16* xrow = xh + (size_t)(m0 + row) * 1024;

  v8f acc[4];
#pragma unroll
  for (int t = 0; t < 4; ++t) zero8(acc[t]);

  for (int k0 = 0; k0 < 1024; k0 += 32) {
    const v16h a = load_a_f16(xrow, kb, k0);
#pragma unroll
    for (int t = 0; t < 4; ++t) {
      const _Float16* wrow = waT + (size_t)(n0 + t * 16 + row) * 1024 + k0 + h16 * 16;
      acc[t] = WMMA_F32_F16(a, load_b16(wrow), acc[t]);
    }
  }

  const int sec = n0 >> 10;          // 0=q, 1=k, 2=v (uniform per wave; n0 is 64-aligned)
  const int h = (n0 & 1023) >> 6;    // head (whole 64-wide group is one head)

#pragma unroll
  for (int t = 0; t < 4; ++t) {
    const float bias = b_attn[n0 + t * 16 + row];
    const int d = t * 16 + row;
#pragma unroll
    for (int r = 0; r < 8; ++r) {
      const int m = m0 + r + 8 * h16;
      const int bb = m >> 10;
      const int s = m & 1023;
      const size_t idx = ((size_t)(bb * 16 + h) * 1024 + s) * 64 + d;
      const float val = acc[t][r] + bias;
      if (sec == 0) {
        qh[idx] = (_Float16)val;
      } else if (sec == 1) {
        kh[idx] = (_Float16)val;
        outp[OUT_K_OFF + idx] = val;
      } else {
        outp[OUT_V_OFF + idx] = val;
        vT[((size_t)(bb * 16 + h) * 64 + d) * 1024 + s] = (_Float16)val;
      }
    }
  }
}

// ---------- kernel 2: flash attention with async double-buffered K tiles ----------
__global__ __launch_bounds__(128) void attn_kernel(const _Float16* __restrict__ qh,
                                                   const _Float16* __restrict__ kh,
                                                   const _Float16* __restrict__ vT,
                                                   _Float16* __restrict__ ah) {
  __shared__ __align__(128) _Float16 ldsK[4 * 2 * 32 * 64];  // 4 waves x 2 bufs x 4KB
  __shared__ __align__(128) _Float16 ldsP[4 * 16 * 32];      // P staging per wave
  const int wave = threadIdx.x >> 5;
  const int lane = threadIdx.x & 31;
  const int gw = blockIdx.x * 4 + wave;   // 0..4095
  const int qt = gw & 63;
  const int h  = (gw >> 6) & 15;
  const int bb = gw >> 10;
  const int m0 = qt * 16;
  const int row = lane & 15;
  const int h16 = lane >> 4;
  const int kb = 8 * h16;

  const _Float16* qrow = qh + (((size_t)(bb * 16 + h)) * 1024 + m0 + row) * 64;
  const _Float16* kbh  = kh + ((size_t)(bb * 16 + h)) * 1024 * 64;
  const _Float16* vTh  = vT + ((size_t)(bb * 16 + h)) * 64 * 1024;
  _Float16* myK = ldsK + wave * (2 * 32 * 64);
  _Float16* myP = ldsP + wave * (16 * 32);
  const unsigned myK_lds = (unsigned)(size_t)myK;   // low 32 bits = LDS byte offset

  const int nblocks = (m0 + 16 + 31) >> 5;          // causal: key blocks of 32
  async_ktile(kbh, 0, lane, myK_lds);               // prefetch block 0

  const v16h qA0 = load_a_f16(qrow, kb, 0);
  const v16h qA1 = load_a_f16(qrow, kb, 32);

  v8f acc[4];
#pragma unroll
  for (int t = 0; t < 4; ++t) zero8(acc[t]);
  float mrun[8], lrun[8];
#pragma unroll
  for (int r = 0; r < 8; ++r) { mrun[r] = -1e30f; lrun[r] = 0.0f; }

  for (int jb = 0; jb < nblocks; ++jb) {
    const int kb0 = jb * 32;

    // prefetch next K tile into the other buffer, then wait for current tile
    if (jb + 1 < nblocks) {
      async_ktile(kbh, (jb + 1) * 32, lane, myK_lds + (unsigned)(((jb + 1) & 1) * 4096));
      asm volatile("s_wait_asynccnt 0x8" ::: "memory");
    } else {
      asm volatile("s_wait_asynccnt 0x0" ::: "memory");
    }
    const _Float16* kt = myK + (jb & 1) * (32 * 64);  // 32 rows x 64 f16 in LDS

    // -------- scores: two 16-key subtiles, K-dim = HD = 64 --------
    v8f s1; zero8(s1);
    v8f s2; zero8(s2);
    {
      const _Float16* kr = kt + row * 64;             // key local row = N
      s1 = WMMA_F32_F16(qA0, load_b16(kr + h16 * 16), s1);
      s1 = WMMA_F32_F16(qA1, load_b16(kr + 32 + h16 * 16), s1);
    }
    {
      const _Float16* kr = kt + (row + 16) * 64;
      s2 = WMMA_F32_F16(qA0, load_b16(kr + h16 * 16), s2);
      s2 = WMMA_F32_F16(qA1, load_b16(kr + 32 + h16 * 16), s2);
    }

    // -------- scale + causal mask + row max --------
    float rm[8];
#pragma unroll
    for (int r = 0; r < 8; ++r) {
      const int q = m0 + r + 8 * h16;
      float v1 = s1[r] * 0.125f;                      // 1/sqrt(64)
      float v2 = s2[r] * 0.125f;
      if (kb0 + row > q)      v1 = -1e30f;
      if (kb0 + 16 + row > q) v2 = -1e30f;
      s1[r] = v1; s2[r] = v2;
      rm[r] = fmaxf(v1, v2);
    }
#pragma unroll
    for (int r = 0; r < 8; ++r) {
      float m = rm[r];
      m = fmaxf(m, __shfl_xor(m, 1, 32));
      m = fmaxf(m, __shfl_xor(m, 2, 32));
      m = fmaxf(m, __shfl_xor(m, 4, 32));
      m = fmaxf(m, __shfl_xor(m, 8, 32));
      rm[r] = m;
    }

    // -------- online softmax update --------
    float alpha[8], rs[8];
#pragma unroll
    for (int r = 0; r < 8; ++r) {
      const float nm = fmaxf(mrun[r], rm[r]);
      alpha[r] = __expf(mrun[r] - nm);
      mrun[r] = nm;
      const float p1 = __expf(s1[r] - nm);
      const float p2 = __expf(s2[r] - nm);
      s1[r] = p1; s2[r] = p2;
      rs[r] = p1 + p2;
    }
#pragma unroll
    for (int r = 0; r < 8; ++r) {
      float t = rs[r];
      t += __shfl_xor(t, 1, 32);
      t += __shfl_xor(t, 2, 32);
      t += __shfl_xor(t, 4, 32);
      t += __shfl_xor(t, 8, 32);
      lrun[r] = lrun[r] * alpha[r] + t;
    }
#pragma unroll
    for (int t = 0; t < 4; ++t)
#pragma unroll
      for (int r = 0; r < 8; ++r) acc[t][r] *= alpha[r];

    // -------- P (D-layout) -> LDS 16x32 row-major -> A-fragment --------
#pragma unroll
    for (int r = 0; r < 8; ++r) {
      const int M = r + 8 * h16;
      myP[M * 32 + row]      = (_Float16)s1[r];
      myP[M * 32 + 16 + row] = (_Float16)s2[r];
    }
    asm volatile("s_wait_dscnt 0x0" ::: "memory");
    const v16h pA = load_a_f16(myP + row * 32, kb, 0);

    // -------- PV: 4 output d-tiles, contiguous vT rows --------
#pragma unroll
    for (int t = 0; t < 4; ++t) {
      const _Float16* vrow = vTh + (size_t)(t * 16 + row) * 1024 + kb0 + h16 * 16;
      acc[t] = WMMA_F32_F16(pA, load_b16(vrow), acc[t]);
    }
  }

  // -------- normalize; write a (B,S,NX) in f16 for the projection --------
#pragma unroll
  for (int t = 0; t < 4; ++t)
#pragma unroll
    for (int r = 0; r < 8; ++r) {
      const float val = acc[t][r] / lrun[r];
      const int s = m0 + r + 8 * h16;
      const int col = h * 64 + t * 16 + row;
      ah[((size_t)(bb * 1024 + s)) * 1024 + col] = (_Float16)val;
    }
}

// ---------- kernel 3: output projection, 16x64 per wave ----------
__global__ __launch_bounds__(256) void proj_kernel(const _Float16* __restrict__ ah,
                                                   const _Float16* __restrict__ wpT,
                                                   const float* __restrict__ b_proj,
                                                   float* __restrict__ out) {
  const int wave = threadIdx.x >> 5;
  const int lane = threadIdx.x & 31;
  const int gw = blockIdx.x * 8 + wave;   // 256 m-tiles x 16 n-groups
  const int NG = 1024 / 64;
  const int tm = gw / NG, tg = gw % NG;
  const int m0 = tm * 16, n0 = tg * 64;
  const int row = lane & 15;
  const int h16 = lane >> 4;
  const int kb = 8 * h16;

  const _Float16* arow = ah + (size_t)(m0 + row) * 1024;

  v8f acc[4];
#pragma unroll
  for (int t = 0; t < 4; ++t) zero8(acc[t]);

  for (int k0 = 0; k0 < 1024; k0 += 32) {
    const v16h a = load_a_f16(arow, kb, k0);
#pragma unroll
    for (int t = 0; t < 4; ++t) {
      const _Float16* wrow = wpT + (size_t)(n0 + t * 16 + row) * 1024 + k0 + h16 * 16;
      acc[t] = WMMA_F32_F16(a, load_b16(wrow), acc[t]);
    }
  }

#pragma unroll
  for (int t = 0; t < 4; ++t) {
    const float bias = b_proj[n0 + t * 16 + row];
#pragma unroll
    for (int r = 0; r < 8; ++r) {
      const int m = m0 + r + 8 * h16;
      out[(size_t)m * 1024 + n0 + t * 16 + row] = acc[t][r] + bias;
    }
  }
}

extern "C" void kernel_launch(void* const* d_in, const int* in_sizes, int n_in,
                              void* d_out, int out_size, void* d_ws, size_t ws_size,
                              hipStream_t stream) {
  const float* x      = (const float*)d_in[0];
  const float* w_attn = (const float*)d_in[1];
  const float* b_attn = (const float*)d_in[2];
  const float* w_proj = (const float*)d_in[3];
  const float* b_proj = (const float*)d_in[4];
  // d_in[5] = causal_bias (mask computed analytically in-kernel)

  float* out = (float*)d_out;
  char* ws = (char*)d_ws;
  _Float16* waT = (_Float16*)(ws + WS_WAT);
  _Float16* wpT = (_Float16*)(ws + WS_WPT);
  _Float16* qh  = (_Float16*)(ws + WS_QH);
  _Float16* kh  = (_Float16*)(ws + WS_KH);
  _Float16* vT  = (_Float16*)(ws + WS_VT);
  _Float16* ah  = (_Float16*)(ws + WS_AH);
  _Float16* xh  = (_Float16*)(ws + WS_XH);

  // 0) f32 -> f16: w_attn^T, w_proj^T, x  (8,388,608 elements)
  cvt_kernel<<<32768, 256, 0, stream>>>(x, w_attn, w_proj, xh, waT, wpT);

  // 1) QKV GEMM: 4096x3072x1024, 16x64 per wave -> 12288 waves
  qkv_kernel<<<12288 / 8, 256, 0, stream>>>(xh, waT, b_attn, qh, kh, vT, out);

  // 2) flash attention: one wave per (b, h, 16-query tile) = 4096 waves
  attn_kernel<<<4096 / 4, 128, 0, stream>>>(qh, kh, vT, ah);

  // 3) output projection: 4096x1024x1024, 16x64 per wave -> 4096 waves
  proj_kernel<<<4096 / 8, 256, 0, stream>>>(ah, wpT, b_proj, out);
}
